// CrossAttentionPro_65094524338743
// MI455X (gfx1250) — matrix operand order repaired and checked by reading, not compiled, not run on hardware
//
#include <hip/hip_runtime.h>
#include <hip/hip_bf16.h>

typedef __bf16 v16bf __attribute__((ext_vector_type(16)));
typedef float  v8f   __attribute__((ext_vector_type(8)));

#define BATCH 8
#define NTOK  1024
#define MTOK  1024
#define CDIM  768
#define HEADS 12
#define DHEAD 64

// ---------------------------------------------------------------------------
// WMMA register-image index math (CDNA5 ISA 7.12.2, wave32, bf16 16x16x32)
// A 16x32: lane = m + 16*((K>>3)&1),  elem = (K&7) + 8*(K>>4)
// B 32x16: lane = n + 16*((K>>3)&1), elem = (K&7) + 8*(K>>4)
// C/D 16x16 f32: row = vgpr + 8*(lane>>4), col = lane&15
// LDS tiles stored as [subtile][lane][16]: fragment = 32 contiguous bytes/lane.
// ---------------------------------------------------------------------------

#define FRAG_DW 1024   // dwords per [4][32] v16bf buffer

__device__ inline unsigned pack2bf(float x, float y) {
    __bf16 a = (__bf16)x, b = (__bf16)y;
    unsigned short ua = __builtin_bit_cast(unsigned short, a);
    unsigned short ub = __builtin_bit_cast(unsigned short, b);
    return (unsigned)ua | ((unsigned)ub << 16);
}

__device__ inline int frag_slot(int sub, int lane, int e) {
    return ((sub * 32 + lane) * 16 + e) >> 1;   // e always even
}
__device__ inline int a_slot(int sub, int m, int K) {
    int lane = m + (((K >> 3) & 1) << 4);
    int e    = (K & 7) + ((K >> 4) << 3);
    return frag_slot(sub, lane, e);
}
__device__ inline int b_slot(int sub, int n, int K) {
    int lane = n + (((K >> 3) & 1) << 4);
    int e    = (K & 7) + ((K >> 4) << 3);
    return frag_slot(sub, lane, e);
}

__device__ inline v8f wmma_bf16(v16bf a, v16bf b, v8f c) {
    return __builtin_amdgcn_wmma_f32_16x16x32_bf16(false, a, false, b, (short)0, c,
                                                   false, false);
}
__device__ inline v8f zero8() { v8f z; for (int i = 0; i < 8; ++i) z[i] = 0.f; return z; }

// ---------------------------------------------------------------------------
// Projection GEMM with double-buffered LDS: C = A[Mrows x K] * Bw[K x Ncols]
// mode 0: q -> bf16 [B,H,N,d] scaled; mode 1: kv -> k/v bf16; mode 2: f32+bias
// block 128 (4 waves); tile 64x64; K step 32
// ---------------------------------------------------------------------------
__global__ void __launch_bounds__(128)
proj_gemm_kernel(const float* __restrict__ A, int lda,
                 const float* __restrict__ Bw, int ldb,
                 int Kdim, int mode,
                 __bf16* __restrict__ outQ, __bf16* __restrict__ outK,
                 __bf16* __restrict__ outV,
                 float* __restrict__ outF, const float* __restrict__ bias,
                 float qscale) {
    __shared__ v16bf Afrag[2][4][32];
    __shared__ v16bf Bfrag[2][4][32];
    unsigned* aw = (unsigned*)Afrag;
    unsigned* bw = (unsigned*)Bfrag;

    const int tid  = threadIdx.x;
    const int wave = tid >> 5;
    const int lane = tid & 31;
    const int rowBase = blockIdx.y * 64;
    const int colBase = blockIdx.x * 64;

    v8f acc[4];
#pragma unroll
    for (int t = 0; t < 4; ++t) acc[t] = zero8();

    float2 aReg[8];
    float  bRegX[8], bRegY[8];

    auto loadTiles = [&](int k0) {
#pragma unroll
        for (int i = 0; i < 8; ++i) {
            int p = i * 128 + tid;
            int r = p >> 4, c = (p & 15) << 1;
            aReg[i] = *(const float2*)(A + (size_t)(rowBase + r) * lda + k0 + c);
        }
#pragma unroll
        for (int i = 0; i < 8; ++i) {
            int p = i * 128 + tid;
            int n = p & 63, k = (p >> 6) << 1;
            bRegX[i] = Bw[(size_t)(k0 + k) * ldb + colBase + n];
            bRegY[i] = Bw[(size_t)(k0 + k + 1) * ldb + colBase + n];
        }
    };
    auto storeTiles = [&](int buf) {
#pragma unroll
        for (int i = 0; i < 8; ++i) {
            int p = i * 128 + tid;
            int r = p >> 4, c = (p & 15) << 1;
            aw[buf * FRAG_DW + a_slot(r >> 4, r & 15, c)] = pack2bf(aReg[i].x, aReg[i].y);
        }
#pragma unroll
        for (int i = 0; i < 8; ++i) {
            int p = i * 128 + tid;
            int n = p & 63, k = (p >> 6) << 1;
            bw[buf * FRAG_DW + b_slot(n >> 4, n & 15, k)] = pack2bf(bRegX[i], bRegY[i]);
        }
    };

    loadTiles(0);
    storeTiles(0);
    __syncthreads();

    const int iters = Kdim >> 5;
    for (int it = 0; it < iters; ++it) {
        const int buf = it & 1;
        const bool more = (it + 1) < iters;
        if (more) loadTiles((it + 1) << 5);

        v16bf a  = Afrag[buf][wave][lane];
        v16bf b0 = Bfrag[buf][0][lane];
        v16bf b1 = Bfrag[buf][1][lane];
        v16bf b2 = Bfrag[buf][2][lane];
        v16bf b3 = Bfrag[buf][3][lane];
        acc[0] = wmma_bf16(a, b0, acc[0]);
        acc[1] = wmma_bf16(a, b1, acc[1]);
        acc[2] = wmma_bf16(a, b2, acc[2]);
        acc[3] = wmma_bf16(a, b3, acc[3]);

        if (more) {
            storeTiles(buf ^ 1);
            __syncthreads();
        }
    }

#pragma unroll
    for (int t = 0; t < 4; ++t) {
#pragma unroll
        for (int r = 0; r < 8; ++r) {
            int row = rowBase + wave * 16 + r + ((lane >> 4) << 3);
            int col = colBase + t * 16 + (lane & 15);
            float v = acc[t][r];
            if (mode == 0) {
                int b = row >> 10, n = row & 1023;
                int h = col >> 6,  dd = col & 63;
                outQ[(((size_t)b * HEADS + h) * NTOK + n) * DHEAD + dd] =
                    (__bf16)(v * qscale);
            } else if (mode == 1) {
                int b = row >> 10, m = row & 1023;
                int kvi = (col >= HEADS * DHEAD) ? 1 : 0;
                int c2  = col - kvi * (HEADS * DHEAD);
                int h   = c2 >> 6, dd = c2 & 63;
                __bf16* dst = kvi ? outV : outK;
                dst[(((size_t)b * HEADS + h) * MTOK + m) * DHEAD + dd] = (__bf16)v;
            } else {
                outF[(size_t)row * CDIM + col] = v + bias[col];
            }
        }
    }
}

// ---------------------------------------------------------------------------
// Per-head QK^T: scores[bh][n][m]; block 128; tile 64x64; K=64 staged once
// ---------------------------------------------------------------------------
__global__ void __launch_bounds__(128)
qk_kernel(const __bf16* __restrict__ q, const __bf16* __restrict__ k,
          __bf16* __restrict__ scores) {
    __shared__ v16bf Qf[2][4][32];   // [kstep][row-subtile][lane]
    __shared__ v16bf Kf[2][4][32];   // [kstep][col-subtile][lane]
    unsigned* qw = (unsigned*)Qf;
    unsigned* kw = (unsigned*)Kf;

    const int tid  = threadIdx.x;
    const int wave = tid >> 5;
    const int lane = tid & 31;
    const int mBase = blockIdx.x * 64;
    const int nBase = blockIdx.y * 64;
    const int bh    = blockIdx.z;

    const __bf16* qp = q + (size_t)bh * NTOK * DHEAD;
    const __bf16* kp = k + (size_t)bh * MTOK * DHEAD;

#pragma unroll
    for (int i = 0; i < 16; ++i) {
        int p = i * 128 + tid;
        int n = p >> 5, dd = (p & 31) << 1;
        unsigned w = *(const unsigned*)(qp + (size_t)(nBase + n) * DHEAD + dd);
        qw[a_slot((dd >> 5) * 4 + (n >> 4), n & 15, dd & 31)] = w;
    }
#pragma unroll
    for (int i = 0; i < 16; ++i) {
        int p = i * 128 + tid;
        int m = p >> 5, dd = (p & 31) << 1;
        unsigned w = *(const unsigned*)(kp + (size_t)(mBase + m) * DHEAD + dd);
        kw[b_slot((dd >> 5) * 4 + (m >> 4), m & 15, dd & 31)] = w;
    }
    __syncthreads();

    v8f acc[4];
#pragma unroll
    for (int t = 0; t < 4; ++t) acc[t] = zero8();

#pragma unroll
    for (int ks = 0; ks < 2; ++ks) {
        v16bf a  = Qf[ks][wave][lane];
        v16bf b0 = Kf[ks][0][lane];
        v16bf b1 = Kf[ks][1][lane];
        v16bf b2 = Kf[ks][2][lane];
        v16bf b3 = Kf[ks][3][lane];
        acc[0] = wmma_bf16(a, b0, acc[0]);
        acc[1] = wmma_bf16(a, b1, acc[1]);
        acc[2] = wmma_bf16(a, b2, acc[2]);
        acc[3] = wmma_bf16(a, b3, acc[3]);
    }

#pragma unroll
    for (int t = 0; t < 4; ++t)
#pragma unroll
        for (int r = 0; r < 8; ++r) {
            int n = nBase + wave * 16 + r + ((lane >> 4) << 3);
            int m = mBase + t * 16 + (lane & 15);
            scores[((size_t)bh * NTOK + n) * MTOK + m] = (__bf16)acc[t][r];
        }
}

// ---------------------------------------------------------------------------
// Row softmax over M (block 256, one row per block), bf16 in/out
// ---------------------------------------------------------------------------
__global__ void __launch_bounds__(256)
softmax_kernel(const __bf16* __restrict__ S, __bf16* __restrict__ P) {
    const int n  = blockIdx.x;
    const int bh = blockIdx.y;
    const int tid = threadIdx.x;
    const __bf16* row = S + ((size_t)bh * NTOK + n) * MTOK;
    __bf16*       out = P + ((size_t)bh * NTOK + n) * MTOK;

    __shared__ float red[8];

    float v[4];
    float mx = -1e30f;
#pragma unroll
    for (int i = 0; i < 4; ++i) {
        v[i] = (float)row[tid + i * 256];
        mx = fmaxf(mx, v[i]);
    }
#pragma unroll
    for (int off = 16; off > 0; off >>= 1) mx = fmaxf(mx, __shfl_xor(mx, off, 32));
    if ((tid & 31) == 0) red[tid >> 5] = mx;
    __syncthreads();
    float bm = red[0];
#pragma unroll
    for (int j = 1; j < 8; ++j) bm = fmaxf(bm, red[j]);
    __syncthreads();

    float s = 0.f;
#pragma unroll
    for (int i = 0; i < 4; ++i) { v[i] = __expf(v[i] - bm); s += v[i]; }
#pragma unroll
    for (int off = 16; off > 0; off >>= 1) s += __shfl_xor(s, off, 32);
    if ((tid & 31) == 0) red[tid >> 5] = s;
    __syncthreads();
    float tot = 0.f;
#pragma unroll
    for (int j = 0; j < 8; ++j) tot += red[j];
    float inv = 1.f / tot;
#pragma unroll
    for (int i = 0; i < 4; ++i) out[tid + i * 256] = (__bf16)(v[i] * inv);
}

// ---------------------------------------------------------------------------
// Depthwise 3x3 conv (SAME, zero pad) + per-head bias on the attention map
// ---------------------------------------------------------------------------
__global__ void __launch_bounds__(256)
dwconv_kernel(const __bf16* __restrict__ P, __bf16* __restrict__ Q,
              const float* __restrict__ w, const float* __restrict__ bias) {
    const int m  = blockIdx.x * 16 + threadIdx.x;
    const int n  = blockIdx.y * 16 + threadIdx.y;
    const int bh = blockIdx.z;
    const int h  = bh % HEADS;
    const __bf16* base = P + (size_t)bh * NTOK * MTOK;
    const float* wh = w + h * 9;

    float acc = bias[h];
#pragma unroll
    for (int dn = 0; dn < 3; ++dn) {
        int nn = n + dn - 1;
        if (nn < 0 || nn >= NTOK) continue;
#pragma unroll
        for (int dm = 0; dm < 3; ++dm) {
            int mm = m + dm - 1;
            if (mm < 0 || mm >= MTOK) continue;
            acc += wh[dn * 3 + dm] * (float)base[(size_t)nn * MTOK + mm];
        }
    }
    Q[(size_t)bh * NTOK * MTOK + (size_t)n * MTOK + m] = (__bf16)acc;
}

// ---------------------------------------------------------------------------
// AV with double-buffered LDS: out[b][n][h*64+d] = sum_m attn * v
// block 128 (4 waves); tile 64(n) x 64(d); K loop over M step 32
// ---------------------------------------------------------------------------
__global__ void __launch_bounds__(128)
av_kernel(const __bf16* __restrict__ attn, const __bf16* __restrict__ v,
          float* __restrict__ out) {
    __shared__ v16bf Pf[2][4][32];
    __shared__ v16bf Vf[2][4][32];
    unsigned* pw = (unsigned*)Pf;
    unsigned* vw = (unsigned*)Vf;

    const int tid  = threadIdx.x;
    const int wave = tid >> 5;
    const int lane = tid & 31;
    const int nBase = blockIdx.x * 64;
    const int bh    = blockIdx.y;
    const int b = bh / HEADS, h = bh % HEADS;

    const __bf16* ap = attn + (size_t)bh * NTOK * MTOK;
    const __bf16* vp = v + (size_t)bh * MTOK * DHEAD;

    v8f acc[4];
#pragma unroll
    for (int t = 0; t < 4; ++t) acc[t] = zero8();

    unsigned pReg[8];
    unsigned v0Reg[4], v1Reg[4];

    auto loadTiles = [&](int m0) {
#pragma unroll
        for (int i = 0; i < 8; ++i) {
            int p = i * 128 + tid;
            int r = p >> 4, c = (p & 15) << 1;
            pReg[i] = *(const unsigned*)(ap + (size_t)(nBase + r) * MTOK + m0 + c);
        }
#pragma unroll
        for (int i = 0; i < 4; ++i) {
            int p = i * 128 + tid;
            int dd = (p & 31) << 1, kk = (p >> 5) << 1;
            v0Reg[i] = *(const unsigned*)(vp + (size_t)(m0 + kk) * DHEAD + dd);
            v1Reg[i] = *(const unsigned*)(vp + (size_t)(m0 + kk + 1) * DHEAD + dd);
        }
    };
    auto storeTiles = [&](int buf) {
#pragma unroll
        for (int i = 0; i < 8; ++i) {
            int p = i * 128 + tid;
            int r = p >> 4, c = (p & 15) << 1;
            pw[buf * FRAG_DW + a_slot(r >> 4, r & 15, c)] = pReg[i];
        }
#pragma unroll
        for (int i = 0; i < 4; ++i) {
            int p = i * 128 + tid;
            int dd = (p & 31) << 1, kk = (p >> 5) << 1;
            unsigned a0 = v0Reg[i], a1 = v1Reg[i];
            unsigned w0 = (a0 & 0xffffu) | (a1 << 16);
            unsigned w1 = (a0 >> 16) | (a1 & 0xffff0000u);
            vw[buf * FRAG_DW + b_slot(dd >> 4, dd & 15, kk)]       = w0;
            vw[buf * FRAG_DW + b_slot(dd >> 4, (dd + 1) & 15, kk)] = w1;
        }
    };

    loadTiles(0);
    storeTiles(0);
    __syncthreads();

    const int iters = MTOK >> 5;   // 32
    for (int it = 0; it < iters; ++it) {
        const int buf = it & 1;
        const bool more = (it + 1) < iters;
        if (more) loadTiles((it + 1) << 5);

        v16bf a  = Pf[buf][wave][lane];
        v16bf b0 = Vf[buf][0][lane];
        v16bf b1 = Vf[buf][1][lane];
        v16bf b2 = Vf[buf][2][lane];
        v16bf b3 = Vf[buf][3][lane];
        acc[0] = wmma_bf16(a, b0, acc[0]);
        acc[1] = wmma_bf16(a, b1, acc[1]);
        acc[2] = wmma_bf16(a, b2, acc[2]);
        acc[3] = wmma_bf16(a, b3, acc[3]);

        if (more) {
            storeTiles(buf ^ 1);
            __syncthreads();
        }
    }

#pragma unroll
    for (int t = 0; t < 4; ++t)
#pragma unroll
        for (int r = 0; r < 8; ++r) {
            int n  = nBase + wave * 16 + r + ((lane >> 4) << 3);
            int dd = t * 16 + (lane & 15);
            out[((size_t)b * NTOK + n) * CDIM + h * DHEAD + dd] = acc[t][r];
        }
}

// ---------------------------------------------------------------------------
extern "C" void kernel_launch(void* const* d_in, const int* in_sizes, int n_in,
                              void* d_out, int out_size, void* d_ws, size_t ws_size,
                              hipStream_t stream) {
    const float* x       = (const float*)d_in[0];
    const float* context = (const float*)d_in[1];
    const float* Wq      = (const float*)d_in[2];
    const float* Wkv     = (const float*)d_in[3];
    const float* conv_w  = (const float*)d_in[4];
    const float* conv_b  = (const float*)d_in[5];
    const float* Wp      = (const float*)d_in[6];
    const float* bp      = (const float*)d_in[7];
    float* out = (float*)d_out;

    const size_t qkvElems = (size_t)BATCH * HEADS * NTOK * DHEAD;
    const size_t mapElems = (size_t)BATCH * HEADS * NTOK * MTOK;
    const size_t aoElems  = (size_t)BATCH * NTOK * CDIM;
    char* p = (char*)d_ws;
    __bf16* qb = (__bf16*)p;  p += qkvElems * sizeof(__bf16);
    __bf16* kb = (__bf16*)p;  p += qkvElems * sizeof(__bf16);
    __bf16* vb = (__bf16*)p;  p += qkvElems * sizeof(__bf16);
    float*  ao = (float*)p;   p += aoElems * sizeof(float);
    __bf16* S1 = (__bf16*)p;  p += mapElems * sizeof(__bf16);  // scores
    __bf16* S2 = (__bf16*)p;                                    // softmax
    __bf16* S3 = S1;   // conv output aliases dead score buffer

    const float qscale = 0.125f;          // 64^-0.5
    const int rows = BATCH * NTOK;        // 8192

    proj_gemm_kernel<<<dim3(CDIM / 64, rows / 64), 128, 0, stream>>>(
        x, CDIM, Wq, CDIM, CDIM, 0, qb, nullptr, nullptr, nullptr, nullptr, qscale);
    proj_gemm_kernel<<<dim3(2 * CDIM / 64, rows / 64), 128, 0, stream>>>(
        context, CDIM, Wkv, 2 * CDIM, CDIM, 1, nullptr, kb, vb, nullptr, nullptr, 1.f);
    qk_kernel<<<dim3(MTOK / 64, NTOK / 64, BATCH * HEADS), 128, 0, stream>>>(qb, kb, S1);
    softmax_kernel<<<dim3(NTOK, BATCH * HEADS), 256, 0, stream>>>(S1, S2);
    dwconv_kernel<<<dim3(MTOK / 16, NTOK / 16, BATCH * HEADS), dim3(16, 16), 0, stream>>>(
        S2, S3, conv_w, conv_b);
    av_kernel<<<dim3(NTOK / 64, BATCH * HEADS), 128, 0, stream>>>(S3, vb, ao);
    proj_gemm_kernel<<<dim3(CDIM / 64, rows / 64), 128, 0, stream>>>(
        ao, CDIM, Wp, CDIM, CDIM, 2, nullptr, nullptr, nullptr, out, bp, 1.f);
}
